// AttentionCombiner_37383395345130
// MI455X (gfx1250) — matrix operand order repaired and checked by reading, not compiled, not run on hardware
//
#include <hip/hip_runtime.h>
#include <hip/hip_bf16.h>

typedef __attribute__((ext_vector_type(16))) __bf16       v16bf;
typedef __attribute__((ext_vector_type(2)))  __bf16       v2bf;
typedef __attribute__((ext_vector_type(8)))  float        v8f;
typedef __attribute__((ext_vector_type(4)))  float        v4f;
typedef __attribute__((ext_vector_type(4)))  unsigned int v4u;
typedef __attribute__((ext_vector_type(2)))  unsigned int v2u;

#define SLEN 2048
#define DIN  512
#define HD   128

__device__ __forceinline__ unsigned short f2bf(float f) {
  unsigned int u = __float_as_uint(f);
  return (unsigned short)((u + 0x7FFFu + ((u >> 16) & 1u)) >> 16);
}
__device__ __forceinline__ unsigned int pkbf(float lo, float hi) {
#if __has_builtin(__builtin_amdgcn_cvt_pk_bf16_f32)
  union { v2bf h; unsigned int u; } cv;
  cv.h = __builtin_amdgcn_cvt_pk_bf16_f32(lo, hi);
  return cv.u;
#else
  return (unsigned int)f2bf(lo) | ((unsigned int)f2bf(hi) << 16);
#endif
}

union Frag {
  v16bf h;
  unsigned int u[8];
  v4u q[2];
};

__device__ __forceinline__ v8f wmma_bf16(const Frag& a, const Frag& b, v8f c) {
  return __builtin_amdgcn_wmma_f32_16x16x32_bf16(false, a.h, false, b.h,
                                                 (short)0, c, false, false);
}

// CDNA5 async global->LDS copy (GV mode: per-lane LDS byte offset + 64b addr).
__device__ __forceinline__ void async_copy_b128(unsigned lds_off, const void* g) {
  asm volatile("global_load_async_to_lds_b128 %0, %1, off"
               :: "v"(lds_off), "v"(g) : "memory");
}
__device__ __forceinline__ void wait_async0() {
  asm volatile("s_wait_asynccnt 0x0" ::: "memory");
}
__device__ __forceinline__ void wait_async4() {  // oldest 4 done (in-order ret)
  asm volatile("s_wait_asynccnt 0x4" ::: "memory");
}
__device__ __forceinline__ unsigned lds_off_of(const void* p) {
  return (unsigned)(unsigned long long)p;  // LDS flat addr: low 32 bits = offset
}

// ---------------------------------------------------------------------------
// Kernel 1: W fp32 [k=1024][n=512] -> bf16, pre-transposed to Wt [n=512][k=1024]
// so the GEMM can async-DMA contiguous [n][k0..k0+32] rows straight into LDS.
// ---------------------------------------------------------------------------
__global__ __launch_bounds__(256) void cvt_w(const float* __restrict__ W,
                                             unsigned short* __restrict__ Wt) {
  int i = blockIdx.x * 256 + threadIdx.x;   // 0 .. 524287
  int n = i & 511;
  int k = i >> 9;
  Wt[(size_t)n * 1024 + k] = f2bf(W[(size_t)k * 512 + n]);
}

// ---------------------------------------------------------------------------
// Kernel 2: flash attention per (n, h, 128-row q tile).
// S^T = K*Q^T so softmaxed P is already in WMMA A-layout in-lane.
// Global->reg stage is software-pipelined one KV tile ahead.
// ---------------------------------------------------------------------------
__global__ __launch_bounds__(256) void flash_kernel(
    const float* __restrict__ o1, const float* __restrict__ o2,
    unsigned short* __restrict__ attnO) {
  __shared__ __align__(16) unsigned short Kt[64][136];  // [token][dim] bf16
  __shared__ __align__(16) unsigned short Vt[128][72];  // [dim][token] bf16

  const int tid = threadIdx.x;
  const int w = tid >> 5;
  const int lane = tid & 31;
  const int ln = lane & 15;
  const int gg = lane >> 4;

  const int bid = blockIdx.x;
  const int qt = bid & 15;
  const int h = (bid >> 4) & 7;
  const int n = bid >> 7;

  const float* Xb = ((h < 4) ? o1 : o2) + (size_t)n * SLEN * DIN + (size_t)(h & 3) * HD;

  // Loop-invariant Q B-fragments: lane = q-row, K(dim) packed along VGPRs.
  Frag qB[4];
  {
    const int qg = qt * 128 + w * 16 + ln;
#pragma unroll
    for (int c = 0; c < 4; ++c) {
      const v4f* qp = (const v4f*)(Xb + (size_t)qg * DIN + c * 32 + gg * 16);
      v4f f0 = qp[0], f1 = qp[1], f2 = qp[2], f3 = qp[3];
      qB[c].u[0] = pkbf(f0[0], f0[1]); qB[c].u[1] = pkbf(f0[2], f0[3]);
      qB[c].u[2] = pkbf(f1[0], f1[1]); qB[c].u[3] = pkbf(f1[2], f1[3]);
      qB[c].u[4] = pkbf(f2[0], f2[1]); qB[c].u[5] = pkbf(f2[2], f2[3]);
      qB[c].u[6] = pkbf(f3[0], f3[1]); qB[c].u[7] = pkbf(f3[2], f3[3]);
    }
  }

  v8f O[8];
#pragma unroll
  for (int i = 0; i < 8; ++i) O[i] = (v8f){};
  float m = -__builtin_inff();
  float l = 0.0f;
  const float KSC = 0.088388347648318447f * 1.4426950408889634f;  // 1/sqrt(128)*log2e

  // Loader mapping: each thread owns tokens tg*4..+3  x  dims dg*8..+7.
  const int tg = tid & 15;   // token group (4 tokens)
  const int dg = tid >> 4;   // dim group (8 dims)

  v4f r[4][2];  // staged tile patch: [token][dim-half]
#pragma unroll
  for (int t = 0; t < 4; ++t) {
    const float* src = Xb + (size_t)(tg * 4 + t) * DIN + dg * 8;
    r[t][0] = ((const v4f*)src)[0];
    r[t][1] = ((const v4f*)src)[1];
  }

  for (int kv0 = 0; kv0 < SLEN; kv0 += 64) {
    // ---- store staged patch to LDS (b128 for Kt, b64 for transposed Vt) ----
#pragma unroll
    for (int t = 0; t < 4; ++t) {
      v4u kp;
      kp[0] = pkbf(r[t][0][0], r[t][0][1]);
      kp[1] = pkbf(r[t][0][2], r[t][0][3]);
      kp[2] = pkbf(r[t][1][0], r[t][1][1]);
      kp[3] = pkbf(r[t][1][2], r[t][1][3]);
      *(v4u*)&Kt[tg * 4 + t][dg * 8] = kp;
    }
#pragma unroll
    for (int e = 0; e < 8; ++e) {
      v2u vp;
      vp[0] = pkbf(r[0][e >> 2][e & 3], r[1][e >> 2][e & 3]);
      vp[1] = pkbf(r[2][e >> 2][e & 3], r[3][e >> 2][e & 3]);
      *(v2u*)&Vt[dg * 8 + e][tg * 4] = vp;
    }
    __syncthreads();

    // ---- issue next tile's global loads early (hide under WMMA work) ----
    v4f rn[4][2];
    if (kv0 + 64 < SLEN) {
#pragma unroll
      for (int t = 0; t < 4; ++t) {
        const float* src = Xb + (size_t)(kv0 + 64 + tg * 4 + t) * DIN + dg * 8;
        if (kv0 + 128 < SLEN) __builtin_prefetch(src + 64 * DIN, 0, 1);
        rn[t][0] = ((const v4f*)src)[0];
        rn[t][1] = ((const v4f*)src)[1];
      }
    }

    // ---- S^T = K x Q^T over 4 token subtiles (preload frags, then WMMA) ----
    v8f st[4];
#pragma unroll
    for (int j = 0; j < 4; ++j) {
      Frag kA[4];
#pragma unroll
      for (int c = 0; c < 4; ++c) {
        kA[c].q[0] = *(const v4u*)&Kt[j * 16 + ln][c * 32 + gg * 16];
        kA[c].q[1] = *(const v4u*)&Kt[j * 16 + ln][c * 32 + gg * 16 + 8];
      }
      v8f s = (v8f){};
#pragma unroll
      for (int c = 0; c < 4; ++c) s = wmma_bf16(kA[c], qB[c], s);
      st[j] = s;
    }

    // ---- online softmax: q-row = lane%16; in-lane reduce + one xor-16 ----
    float mt = -__builtin_inff();
#pragma unroll
    for (int j = 0; j < 4; ++j)
#pragma unroll
      for (int rr = 0; rr < 8; ++rr) {
        st[j][rr] *= KSC;
        mt = fmaxf(mt, st[j][rr]);
      }
    mt = fmaxf(mt, __shfl_xor(mt, 16, 32));
    const float mnew = fmaxf(m, mt);
    const float alpha = exp2f(m - mnew);

    float rs = 0.0f;
    Frag pA[2];  // S^T accumulator layout == A-layout of P (in-lane convert)
#pragma unroll
    for (int tc = 0; tc < 2; ++tc)
#pragma unroll
      for (int v = 0; v < 8; ++v) {
        const int j = tc * 2 + (v >> 2);
        const int rr = (v & 3) * 2;
        float p0 = exp2f(st[j][rr] - mnew);
        float p1 = exp2f(st[j][rr + 1] - mnew);
        rs += p0 + p1;
        pA[tc].u[v] = pkbf(p0, p1);
      }
    rs += __shfl_xor(rs, 16, 32);
    l = l * alpha + rs;
    m = mnew;

    float ab[8];
#pragma unroll
    for (int rr = 0; rr < 8; ++rr) ab[rr] = __shfl(alpha, gg * 8 + rr, 32);

    // ---- O = diag(alpha)*O + P*V (preload both V frags per 16-dim block) ----
#pragma unroll
    for (int db = 0; db < 8; ++db) {
      Frag vB[2];
      const int d = db * 16 + ln;
#pragma unroll
      for (int tc = 0; tc < 2; ++tc) {
        vB[tc].q[0] = *(const v4u*)&Vt[d][tc * 32 + gg * 8];
        vB[tc].q[1] = *(const v4u*)&Vt[d][tc * 32 + 16 + gg * 8];
      }
      v8f o = O[db];
#pragma unroll
      for (int rr = 0; rr < 8; ++rr) o[rr] *= ab[rr];
      o = wmma_bf16(pA[0], vB[0], o);
      o = wmma_bf16(pA[1], vB[1], o);
      O[db] = o;
    }
    __syncthreads();

#pragma unroll
    for (int t = 0; t < 4; ++t) {
      r[t][0] = rn[t][0];
      r[t][1] = rn[t][1];
    }
  }

  // ---- normalize and store bf16 attn output [N*S, H*128] ----
  float li[8];
#pragma unroll
  for (int rr = 0; rr < 8; ++rr) li[rr] = 1.0f / __shfl(l, gg * 8 + rr, 32);

  const size_t rowBase = (size_t)n * SLEN + qt * 128 + w * 16 + gg * 8;
#pragma unroll
  for (int db = 0; db < 8; ++db) {
    const int col = h * HD + db * 16 + ln;
#pragma unroll
    for (int rr = 0; rr < 8; ++rr)
      attnO[(rowBase + rr) * 1024 + col] = f2bf(O[db][rr] * li[rr]);
  }
}

// ---------------------------------------------------------------------------
// Kernel 3: C[8192,512] = A[8192,1024](bf16) @ W + b, W given pre-transposed
// as Wt[n=512][k=1024] bf16. Double-buffered async global->LDS DMA pipeline:
// issue next tile's 4 copies, then s_wait_asynccnt<=4 proves the current
// buffer's copies (issued earlier, in-order completion) have landed.
// ---------------------------------------------------------------------------
__global__ __launch_bounds__(256) void gemm_kernel(
    const unsigned short* __restrict__ A, const unsigned short* __restrict__ Wt,
    const float* __restrict__ bias, float* __restrict__ out) {
  __shared__ __align__(16) unsigned short As[2][128][40];  // [buf][row][k]
  __shared__ __align__(16) unsigned short Ws[2][128][40];  // [buf][n][k]

  const int tid = threadIdx.x;
  const int w = tid >> 5, lane = tid & 31, ln = lane & 15, gg = lane >> 4;
  const int m0 = blockIdx.x * 128;
  const int n0 = blockIdx.y * 128;

  const int row = tid >> 1, half = tid & 1;  // loader mapping (both tiles)
  unsigned asO0[2], asO1[2], wsO0[2], wsO1[2];
#pragma unroll
  for (int b = 0; b < 2; ++b) {
    asO0[b] = lds_off_of(&As[b][row][half * 16]);
    asO1[b] = lds_off_of(&As[b][row][half * 16 + 8]);
    wsO0[b] = lds_off_of(&Ws[b][row][half * 16]);
    wsO1[b] = lds_off_of(&Ws[b][row][half * 16 + 8]);
  }
  const unsigned short* aRow = A + (size_t)(m0 + row) * 1024 + half * 16;
  const unsigned short* wRow = Wt + (size_t)(n0 + row) * 1024 + half * 16;

  v8f acc[8];
#pragma unroll
  for (int i = 0; i < 8; ++i) acc[i] = (v8f){};

  // prologue: DMA first tile into buffer 0
  async_copy_b128(asO0[0], aRow);
  async_copy_b128(asO1[0], aRow + 8);
  async_copy_b128(wsO0[0], wRow);
  async_copy_b128(wsO1[0], wRow + 8);

  int pb = 0;
  for (int k0 = 0; k0 < 1024; k0 += 32, pb ^= 1) {
    const bool more = (k0 + 32) < 1024;
    if (more) {  // uniform branch: EXEC untouched
      const int nb = pb ^ 1;
      async_copy_b128(asO0[nb], aRow + k0 + 32);
      async_copy_b128(asO1[nb], aRow + k0 + 40);
      async_copy_b128(wsO0[nb], wRow + k0 + 32);
      async_copy_b128(wsO1[nb], wRow + k0 + 40);
      wait_async4();   // current buffer's 4 copies complete; next 4 in flight
    } else {
      wait_async0();
    }
    __syncthreads();

    const unsigned short (*Asb)[40] = As[pb];
    const unsigned short (*Wsb)[40] = Ws[pb];
    Frag aF;
    aF.q[0] = *(const v4u*)&Asb[w * 16 + ln][gg * 8];
    aF.q[1] = *(const v4u*)&Asb[w * 16 + ln][16 + gg * 8];
    Frag wF[8];
#pragma unroll
    for (int db = 0; db < 8; ++db) {
      wF[db].q[0] = *(const v4u*)&Wsb[db * 16 + ln][gg * 8];
      wF[db].q[1] = *(const v4u*)&Wsb[db * 16 + ln][16 + gg * 8];
    }
#pragma unroll
    for (int db = 0; db < 8; ++db) acc[db] = wmma_bf16(aF, wF[db], acc[db]);
    __syncthreads();  // all waves done reading buf pb before it is re-DMA'd
  }

#pragma unroll
  for (int db = 0; db < 8; ++db) {
    const int nn = n0 + db * 16 + ln;
    const float bv = bias[nn];
#pragma unroll
    for (int rr = 0; rr < 8; ++rr)
      out[(size_t)(m0 + w * 16 + gg * 8 + rr) * 512 + nn] = acc[db][rr] + bv;
  }
}

// ---------------------------------------------------------------------------
extern "C" void kernel_launch(void* const* d_in, const int* in_sizes, int n_in,
                              void* d_out, int out_size, void* d_ws, size_t ws_size,
                              hipStream_t stream) {
  (void)in_sizes; (void)n_in; (void)out_size; (void)ws_size;
  const float* o1 = (const float*)d_in[0];
  const float* o2 = (const float*)d_in[1];
  const float* Wf = (const float*)d_in[2];
  const float* bf = (const float*)d_in[3];

  unsigned short* attnO = (unsigned short*)d_ws;  // 8192*1024 bf16 = 16 MiB
  unsigned short* Wt = (unsigned short*)((char*)d_ws + (size_t)16 * 1024 * 1024);

  cvt_w<<<2048, 256, 0, stream>>>(Wf, Wt);               // 1024*512 elements
  flash_kernel<<<512, 256, 0, stream>>>(o1, o2, attnO);  // N*H*(S/128)
  gemm_kernel<<<dim3(64, 4), 256, 0, stream>>>(attnO, Wt, bf, (float*)d_out);
}
